// VectorQuantizer_66838281061253
// MI455X (gfx1250) — matrix operand order repaired and checked by reading, not compiled, not run on hardware
//
#include <hip/hip_runtime.h>
#include <hip/hip_bf16.h>

typedef __attribute__((ext_vector_type(16))) __bf16 v16bf;
typedef __attribute__((ext_vector_type(8)))  float  v8f;

#define NUM_K 1024
#define DIM   256
#define NPTS  32768
#define LDS_STRIDE 264   // 256 + 8 ushorts: 132-dword row stride -> bank offset 4 per row

__device__ __forceinline__ unsigned short f2bf_rne(float x) {
  unsigned u = __float_as_uint(x);
  unsigned r = ((u >> 16) & 1u) + 0x7FFFu;
  return (unsigned short)((u + r) >> 16);
}
__device__ __forceinline__ float bf2f(unsigned short h) {
  return __uint_as_float(((unsigned)h) << 16);
}

union Frag16 { uint4 q[2]; v16bf v; };

// ---------------------------------------------------------------------------
// Kernel 1: codebook -> hi/lo bf16 in B-fragment (register) layout + ||e_k||^2
// B order used here (contiguous K, same as A): lane = 16*(c_half) + (k%16); VGPR v holds K pair
// (2v, 2v+1) within the lane's 16-wide K half (c_half = (c%32)/16).
// Fragment buffer: [k_tile(64)][c_step(8)][lane(32)][8 dwords] of bf16 pairs.
// ---------------------------------------------------------------------------
__global__ void vq_prep(const float* __restrict__ cb,
                        unsigned short* __restrict__ fh,
                        unsigned short* __restrict__ fl,
                        float* __restrict__ cbnorm) {
  int k = blockIdx.x, c = threadIdx.x;
  float x = cb[k * DIM + c];
  unsigned short h = f2bf_rne(x);
  unsigned short l = f2bf_rne(x - bf2f(h));

  int ktile = k >> 4, n = k & 15;
  int cstep = c >> 5, cl = c & 31, half = cl >> 4, c16 = cl & 15;
  int v = c16 >> 1, e = c16 & 1;
  int lane = half * 16 + n;
  int pos = (((ktile * 8 + cstep) * 32 + lane) * 8 + v) * 2 + e;
  fh[pos] = h;
  fl[pos] = l;

  __shared__ float red[256];
  red[c] = x * x;
  __syncthreads();
  for (int s = 128; s > 0; s >>= 1) {
    if (c < s) red[c] += red[c + s];
    __syncthreads();
  }
  if (c == 0) cbnorm[k] = red[0];
}

// ---------------------------------------------------------------------------
// Kernel 2: fused distance-GEMM + argmin.
// Block: 32 rows of z (all 256 dims) staged to LDS as hi/lo bf16.
// 8 waves = 2 M-subtiles x 4 K-chunks of 256 codewords.
// Per k-tile: 8 c-steps x 3 bf16 WMMA (hi*hi + hi*lo + lo*hi) ~ fp32 accuracy.
// ---------------------------------------------------------------------------
__global__ void __launch_bounds__(256)
vq_gemm_argmin(const float* __restrict__ z,
               const unsigned short* __restrict__ fh,
               const unsigned short* __restrict__ fl,
               const float* __restrict__ cbnorm,
               int* __restrict__ idx_out) {
  __shared__ unsigned short zh[32][LDS_STRIDE];
  __shared__ unsigned short zl[32][LDS_STRIDE];
  __shared__ float wval[8][16];
  __shared__ int   widx[8][16];

  int tid = threadIdx.x;
  int n0 = blockIdx.x * 32;          // 32 consecutive spatial positions
  int b = n0 >> 10, hw = n0 & 1023;  // hw == h*32 (n0 is 32-aligned)

  // Stage z tile: coalesced 128B reads across w for each channel c.
  {
    int w = tid & 31, cg = tid >> 5;
    const float* zp = z + ((size_t)b * DIM) * 1024 + hw + w;
    for (int c = cg; c < DIM; c += 8) {
      float x = zp[(size_t)c * 1024];
      unsigned short h = f2bf_rne(x);
      zh[w][c] = h;
      zl[w][c] = f2bf_rne(x - bf2f(h));
    }
  }
  __syncthreads();

  int wave = tid >> 5, lane = tid & 31;
  int m_sub = wave & 1, k_chunk = wave >> 1;
  int m = m_sub * 16 + (lane & 15);
  int kh = lane >> 4;  // A order used here (contiguous, as B): lanes 0-15 hold K {0-15}, 16-31 hold {16-31}

  // Load A fragments (hi & lo) for all 8 c-steps into registers.
  v16bf a_hi[8], a_lo[8];
#pragma unroll
  for (int cs = 0; cs < 8; ++cs) {
    int base = cs * 32 + kh * 16;
    Frag16 t;
    t.q[0] = *(const uint4*)&zh[m][base];
    t.q[1] = *(const uint4*)&zh[m][base + 8];
    a_hi[cs] = t.v;
    t.q[0] = *(const uint4*)&zl[m][base];
    t.q[1] = *(const uint4*)&zl[m][base + 8];
    a_lo[cs] = t.v;
  }

  float bestd[8];
  int   bestk[8];
#pragma unroll
  for (int j = 0; j < 8; ++j) { bestd[j] = 3.4e38f; bestk[j] = 0; }

  for (int kt = 0; kt < 16; ++kt) {
    int gt = k_chunk * 16 + kt;  // global 16-codeword tile
    v8f acc = {};
#pragma unroll
    for (int cs = 0; cs < 8; ++cs) {
      size_t off = (size_t)((gt * 8 + cs) * 32 + lane) * 16;  // 16 ushorts/lane
      Frag16 bh, bl;
      const uint4* ph = (const uint4*)(fh + off);
      bh.q[0] = ph[0]; bh.q[1] = ph[1];
      const uint4* pl = (const uint4*)(fl + off);
      bl.q[0] = pl[0]; bl.q[1] = pl[1];
      acc = __builtin_amdgcn_wmma_f32_16x16x32_bf16(false, a_hi[cs], false, bh.v,
                                                    (short)0, acc, false, false);
      acc = __builtin_amdgcn_wmma_f32_16x16x32_bf16(false, a_lo[cs], false, bh.v,
                                                    (short)0, acc, false, false);
      acc = __builtin_amdgcn_wmma_f32_16x16x32_bf16(false, a_hi[cs], false, bl.v,
                                                    (short)0, acc, false, false);
    }
    // Column (codeword) for this lane: N = lane%16.
    int k = gt * 16 + (lane & 15);
    float en = cbnorm[k];
#pragma unroll
    for (int j = 0; j < 8; ++j) {
      float d = en - 2.0f * acc[j];   // ||e||^2 - 2 z.e  (||z||^2 dropped)
      if (d < bestd[j]) { bestd[j] = d; bestk[j] = k; }
    }
  }

  // Reduce across the 16 column-lanes (xor masks <=8 stay within 16-lane halves).
#pragma unroll
  for (int off = 1; off < 16; off <<= 1) {
#pragma unroll
    for (int j = 0; j < 8; ++j) {
      float od = __shfl_xor(bestd[j], off, 32);
      int   ok = __shfl_xor(bestk[j], off, 32);
      if (od < bestd[j] || (od == bestd[j] && ok < bestk[j])) {
        bestd[j] = od; bestk[j] = ok;
      }
    }
  }

  // Row for (j, lane): M = j + 8*(lane>>4). One lane per half writes results.
  if ((lane & 15) == 0) {
#pragma unroll
    for (int j = 0; j < 8; ++j) {
      wval[wave][j + 8 * kh] = bestd[j];
      widx[wave][j + 8 * kh] = bestk[j];
    }
  }
  __syncthreads();

  // Combine the 4 K-chunks per M-subtile; tie-break on smaller k (first argmin).
  if (tid < 32) {
    int r = tid, ms = r >> 4, rl = r & 15;
    float bd = wval[ms][rl];
    int   bk = widx[ms][rl];
#pragma unroll
    for (int cc = 1; cc < 4; ++cc) {
      float od = wval[ms + 2 * cc][rl];
      int   ok = widx[ms + 2 * cc][rl];
      if (od < bd || (od == bd && ok < bk)) { bd = od; bk = ok; }
    }
    idx_out[n0 + r] = bk;
  }
}

// ---------------------------------------------------------------------------
// Kernel 3: one-hot (N x K) — the real bandwidth cost; float4 streaming stores.
// ---------------------------------------------------------------------------
__global__ void vq_onehot(const int* __restrict__ idx, float* __restrict__ out) {
  int n = blockIdx.x;
  int kidx = idx[n];
  float4 v = make_float4(0.f, 0.f, 0.f, 0.f);
  if ((kidx >> 2) == (int)threadIdx.x) ((float*)&v)[kidx & 3] = 1.0f;
  ((float4*)(out + (size_t)n * NUM_K))[threadIdx.x] = v;
}

// ---------------------------------------------------------------------------
// Kernel 4: z_quantized gather, written directly in (B,C,H,W) layout.
// Stores coalesced across w; codebook rows hit L2.
// ---------------------------------------------------------------------------
__global__ void vq_gather(const int* __restrict__ idx,
                          const float* __restrict__ cb,
                          float* __restrict__ outq) {
  int bh = blockIdx.x;                 // b*32 + h
  int b = bh >> 5, hw = (bh & 31) * 32;
  int w = threadIdx.x & 31, cg = threadIdx.x >> 5;
  int kk = idx[bh * 32 + w];
  const float* crow = cb + (size_t)kk * DIM;
  float* op = outq + ((size_t)b * DIM) * 1024 + hw + w;
  for (int c = cg; c < DIM; c += 8) op[(size_t)c * 1024] = crow[c];
}

extern "C" void kernel_launch(void* const* d_in, const int* in_sizes, int n_in,
                              void* d_out, int out_size, void* d_ws, size_t ws_size,
                              hipStream_t stream) {
  const float* z  = (const float*)d_in[0];
  const float* cb = (const float*)d_in[1];
  float* out = (float*)d_out;

  // Workspace: cb_frag_hi (512KB) | cb_frag_lo (512KB) | cbnorm (4KB) | idx (128KB)
  unsigned short* fh = (unsigned short*)d_ws;
  unsigned short* fl = fh + (size_t)NUM_K * DIM;
  float* cbnorm = (float*)(fl + (size_t)NUM_K * DIM);
  int* idx = (int*)(cbnorm + NUM_K);

  vq_prep<<<NUM_K, DIM, 0, stream>>>(cb, fh, fl, cbnorm);
  vq_gemm_argmin<<<NPTS / 32, 256, 0, stream>>>(z, fh, fl, cbnorm, idx);
  vq_onehot<<<NPTS, 256, 0, stream>>>(idx, out);
  vq_gather<<<NPTS / 32, 256, 0, stream>>>(idx, cb, out + (size_t)NPTS * NUM_K);
}
